// FakeNewsModelGated_1408749273892
// MI455X (gfx1250) — matrix operand order repaired and checked by reading, not compiled, not run on hardware
//
#include <hip/hip_runtime.h>

// ---------------------------------------------------------------------------
// CDNA5 (gfx1250) wave32 WMMA types
// ---------------------------------------------------------------------------
typedef __attribute__((ext_vector_type(16))) __bf16 v16bf;
typedef __attribute__((ext_vector_type(8)))  float  v8f;
typedef int v4i __attribute__((vector_size(16)));

// Async global->LDS staging (gfx1250 GLOBAL_LOAD_ASYNC_TO_LDS_B128, ASYNCcnt)
#if defined(__has_builtin)
#if __has_builtin(__builtin_amdgcn_global_load_async_to_lds_b128)
#define USE_ASYNC_LDS 1
#endif
#endif
#ifndef USE_ASYNC_LDS
#define USE_ASYNC_LDS 0
#endif

__device__ __forceinline__ void stage16(const __bf16* g, __bf16* l) {
#if USE_ASYNC_LDS
  __builtin_amdgcn_global_load_async_to_lds_b128(
      (__attribute__((address_space(1))) v4i*)g,
      (__attribute__((address_space(3))) v4i*)l, 0, 0);
#else
  *(uint4*)l = *(const uint4*)g;
#endif
}

__device__ __forceinline__ void stage_wait() {
#if USE_ASYNC_LDS
#if __has_builtin(__builtin_amdgcn_s_wait_asynccnt)
  __builtin_amdgcn_s_wait_asynccnt(0);
#else
  asm volatile("s_wait_asynccnt 0x0" ::: "memory");
#endif
#endif
}

__device__ __forceinline__ unsigned short f2bf_bits(float f) {
  unsigned u = __builtin_bit_cast(unsigned, f);
  unsigned r = u + 0x7fffu + ((u >> 16) & 1u);   // round-to-nearest-even
  return (unsigned short)(r >> 16);
}

__device__ __forceinline__ v8f wmma_bf16(v16bf a, v16bf b, v8f c) {
  // D = A(16x32 bf16) x B(32x16 bf16) + C(16x16 f32)
  return __builtin_amdgcn_wmma_f32_16x16x32_bf16(
      /*neg_a=*/false, a, /*neg_b=*/false, b,
      /*c_mod=*/(short)0, c, /*reuse_a=*/false, /*reuse_b=*/false);
}

// A fragment: 16(M) x 32(K) bf16 tile at `base` (row-major, leading dim `ld`).
// ISA layout: lanes 0-15 hold M=lane, K={kb..kb+7, kb+16..kb+23}, kb=8*(lane>>4).
__device__ __forceinline__ v16bf load_a_frag(const __bf16* base, int ld) {
  const int lane = threadIdx.x & 31;
  const int m  = lane & 15;
  const int kb = (lane >> 4) << 3;            // 0 or 8
  const __bf16* p = base + (size_t)m * ld + kb;
  union { v16bf v; uint4 q[2]; } u;
  u.q[0] = *(const uint4*)(p);                // K = kb .. kb+7
  u.q[1] = *(const uint4*)(p + 16);           // K = kb+16 .. kb+23
  return u.v;
}

// B fragment from W^T convention: B[k][n] = W[n][k], W row-major [16 rows, ld].
// ISA layout: lane n=lane&15 holds column n, K={kb..kb+7, kb+16..kb+23}, kb=8*(lane>>4).
__device__ __forceinline__ v16bf load_b_frag(const __bf16* base, int ld) {
  const int lane = threadIdx.x & 31;
  const int n  = lane & 15;
  const int kb = (lane >> 4) << 3;            // 0 or 8
  const __bf16* p = base + (size_t)n * ld + kb;
  union { v16bf v; uint4 q[2]; } u;
  u.q[0] = *(const uint4*)(p);
  u.q[1] = *(const uint4*)(p + 16);
  return u.v;
}

// ---------------------------------------------------------------------------
// fp32 -> bf16 conversion (vectorized x4)
// ---------------------------------------------------------------------------
__global__ void cvt_f32_bf16(const float* __restrict__ src,
                             __bf16* __restrict__ dst, int n4) {
  int i = blockIdx.x * blockDim.x + threadIdx.x;
  if (i >= n4) return;
  float4 v = ((const float4*)src)[i];
  uint2 o;
  o.x = (unsigned)f2bf_bits(v.x) | ((unsigned)f2bf_bits(v.y) << 16);
  o.y = (unsigned)f2bf_bits(v.z) | ((unsigned)f2bf_bits(v.w) << 16);
  ((uint2*)dst)[i] = o;
}

// ggc_weight [3][512][512] applied as x@W  ->  store transposed bf16 copy
// so every GEMM uses the W[out,in] (W^T) convention.
__global__ void transpose_cvt_ggc(const float* __restrict__ src,
                                  __bf16* __restrict__ dst) {
  int i = blockIdx.x * blockDim.x + threadIdx.x;   // over 3*512*512
  if (i >= 3 * 512 * 512) return;
  int l = i >> 18;
  int r = i & ((1 << 18) - 1);
  int j = r >> 9;           // out row (output channel)
  int k = r & 511;          // out col (input channel)
  float v = src[((size_t)l << 18) + ((size_t)k << 9) + j];
  dst[i] = __builtin_bit_cast(__bf16, f2bf_bits(v));
}

__global__ void zero_f32_4(float4* __restrict__ p, int n4) {
  int i = blockIdx.x * blockDim.x + threadIdx.x;
  if (i < n4) p[i] = make_float4(0.f, 0.f, 0.f, 0.f);
}

// ---------------------------------------------------------------------------
// Tiled GEMM: Y[:, joff:joff+Nout] = A(bf16 [M,K]) @ W(bf16 [Nout,K])^T + bias
// Workgroup = 256 threads (8 waves), tile 128(M) x 128(N), K-step 32 via LDS
// (async staging). Wave (wm,wn) computes a 32x64 block = 8 WMMA accumulators.
// Requires Nout % 128 == 0 (true for 256 and 512 here).
// ---------------------------------------------------------------------------
__global__ __launch_bounds__(256)
void gemm_bf16_wT(const __bf16* __restrict__ A, const __bf16* __restrict__ W,
                  const float* __restrict__ bias, float* __restrict__ Y,
                  int M, int K, int ldY, int joff) {
  __shared__ __bf16 As[128 * 32];
  __shared__ __bf16 Bs[128 * 32];

  const int i0 = blockIdx.x * 128;
  const int j0 = blockIdx.y * 128;
  const int wv = threadIdx.x >> 5;
  const int wm = wv & 3;          // 32-row block within tile
  const int wn = wv >> 2;         // 64-col block within tile
  const int lane = threadIdx.x & 31;

  v8f vz = {};
  v8f acc[2][4];
#pragma unroll
  for (int ti = 0; ti < 2; ++ti)
#pragma unroll
    for (int tj = 0; tj < 4; ++tj) acc[ti][tj] = vz;

  // each thread stages one 16B chunk of A and one of B per K-step
  const int crow  = threadIdx.x >> 1;          // 0..127
  const int chalf = (threadIdx.x & 1) << 4;    // 0 or 16 elements
  const int gr = i0 + crow;
  const int ar = (gr < M) ? gr : (M - 1);      // clamp: dup rows feed only
                                               // never-stored output rows
  const __bf16* ga = A + (size_t)ar * K + chalf;
  const __bf16* gb = W + (size_t)(j0 + crow) * K + chalf;
  __bf16* la = As + crow * 32 + chalf;
  __bf16* lb = Bs + crow * 32 + chalf;

  for (int k0 = 0; k0 < K; k0 += 32) {
    stage16(ga + k0, la);
    stage16(gb + k0, lb);
    stage_wait();
    __syncthreads();

    v16bf af0 = load_a_frag(As + (wm * 32 +  0) * 32, 32);
    v16bf af1 = load_a_frag(As + (wm * 32 + 16) * 32, 32);
#pragma unroll
    for (int tj = 0; tj < 4; ++tj) {
      v16bf bfr = load_b_frag(Bs + (wn * 64 + tj * 16) * 32, 32);
      acc[0][tj] = wmma_bf16(af0, bfr, acc[0][tj]);
      acc[1][tj] = wmma_bf16(af1, bfr, acc[1][tj]);
    }
    __syncthreads();
  }

  // C/D layout: VGPR r -> M = r + 8*(lane>>4); N = lane&15
  const int cn = lane & 15, chi = (lane >> 4) << 3;
#pragma unroll
  for (int ti = 0; ti < 2; ++ti) {
#pragma unroll
    for (int tj = 0; tj < 4; ++tj) {
      int col = j0 + wn * 64 + tj * 16 + cn;
      float bv = bias ? bias[col] : 0.0f;
      v8f a = acc[ti][tj];
#pragma unroll
      for (int r = 0; r < 8; ++r) {
        int row = i0 + wm * 32 + ti * 16 + r + chi;
        if (row < M) Y[(size_t)row * ldY + joff + col] = a[r] + bv;
      }
    }
  }
}

// ---------------------------------------------------------------------------
// Edge scatter-add: agg[dst] += m[src]  (fp32, float4-chunked atomics)
// ---------------------------------------------------------------------------
__global__ void scatter_add_edges(const float* __restrict__ m,
                                  const int* __restrict__ ei,
                                  float* __restrict__ agg, int E) {
  long idx = (long)blockIdx.x * blockDim.x + threadIdx.x;
  if (idx >= (long)E * 128) return;          // 512/4 chunks per edge
  int e = (int)(idx >> 7);
  int c = ((int)idx & 127) << 2;
  int s = ei[e];
  int d = ei[E + e];
  float4 v = *(const float4*)(m + (size_t)s * 512 + c);
  float* ap = agg + (size_t)d * 512 + c;
  atomicAdd(ap + 0, v.x);
  atomicAdd(ap + 1, v.y);
  atomicAdd(ap + 2, v.z);
  atomicAdd(ap + 3, v.w);
}

// ---------------------------------------------------------------------------
// Fused GRU cell: one wave per 16x16 tile of h_new.
// 6 accumulators (i_r,i_z,i_n from agg@W_ih^T; h_r,h_z,h_n from h@W_hh^T),
// 6 WMMAs per K-step from L2-resident operands, then gates + in-place update.
// ---------------------------------------------------------------------------
__global__ __launch_bounds__(32)
void gru_fused(const __bf16* __restrict__ aggb, const __bf16* __restrict__ hb,
               float* __restrict__ h,
               const __bf16* __restrict__ Wih, const __bf16* __restrict__ Whh,
               const float* __restrict__ bih, const float* __restrict__ bhh) {
  const int i0 = blockIdx.x * 16;
  const int j0 = blockIdx.y * 16;
  v8f ir = {}, iz = {}, in_ = {}, hr = {}, hz = {}, hn = {};

  for (int k0 = 0; k0 < 512; k0 += 32) {
    v16bf aa = load_a_frag(aggb + (size_t)i0 * 512 + k0, 512);
    v16bf ah = load_a_frag(hb   + (size_t)i0 * 512 + k0, 512);
    v16bf b0 = load_b_frag(Wih + (size_t)(j0 +    0) * 512 + k0, 512);
    v16bf b1 = load_b_frag(Wih + (size_t)(j0 +  512) * 512 + k0, 512);
    v16bf b2 = load_b_frag(Wih + (size_t)(j0 + 1024) * 512 + k0, 512);
    ir  = wmma_bf16(aa, b0, ir);
    iz  = wmma_bf16(aa, b1, iz);
    in_ = wmma_bf16(aa, b2, in_);
    v16bf c0 = load_b_frag(Whh + (size_t)(j0 +    0) * 512 + k0, 512);
    v16bf c1 = load_b_frag(Whh + (size_t)(j0 +  512) * 512 + k0, 512);
    v16bf c2 = load_b_frag(Whh + (size_t)(j0 + 1024) * 512 + k0, 512);
    hr = wmma_bf16(ah, c0, hr);
    hz = wmma_bf16(ah, c1, hz);
    hn = wmma_bf16(ah, c2, hn);
  }

  const int lane = threadIdx.x & 31;
  const int cn = lane & 15, chi = (lane >> 4) << 3;
  const int col = j0 + cn;
  const float bir = bih[col], biz = bih[512 + col], bin = bih[1024 + col];
  const float bhr = bhh[col], bhz = bhh[512 + col], bhn = bhh[1024 + col];
#pragma unroll
  for (int r = 0; r < 8; ++r) {
    int row = i0 + r + chi;
    float rg = 1.0f / (1.0f + __expf(-(ir[r] + bir + hr[r] + bhr)));
    float zg = 1.0f / (1.0f + __expf(-(iz[r] + biz + hz[r] + bhz)));
    float ng = tanhf(in_[r] + bin + rg * (hn[r] + bhn));
    size_t idx = (size_t)row * 512 + col;
    h[idx] = (1.0f - zg) * ng + zg * h[idx];
  }
}

// ---------------------------------------------------------------------------
// Output head: out[i,:] = relu(h[i,:]) @ W_out^T + b_out   (C=2, trivial)
// ---------------------------------------------------------------------------
__global__ void out_head(const float* __restrict__ h,
                         const float* __restrict__ Wout,
                         const float* __restrict__ bout,
                         float* __restrict__ out, int N) {
  int i = blockIdx.x * blockDim.x + threadIdx.x;
  if (i >= N) return;
  const float4* hp = (const float4*)(h + (size_t)i * 512);
  const float4* w0 = (const float4*)(Wout);
  const float4* w1 = (const float4*)(Wout + 512);
  float a0 = 0.f, a1 = 0.f;
  for (int k = 0; k < 128; ++k) {
    float4 x = hp[k];
    x.x = fmaxf(x.x, 0.f); x.y = fmaxf(x.y, 0.f);
    x.z = fmaxf(x.z, 0.f); x.w = fmaxf(x.w, 0.f);
    float4 u = w0[k], v = w1[k];
    a0 += x.x * u.x + x.y * u.y + x.z * u.z + x.w * u.w;
    a1 += x.x * v.x + x.y * v.y + x.z * v.z + x.w * v.w;
  }
  out[2 * i + 0] = a0 + bout[0];
  out[2 * i + 1] = a1 + bout[1];
}

// ---------------------------------------------------------------------------
// Host-side launcher (graph-capture safe: only kernel launches on `stream`)
// ---------------------------------------------------------------------------
extern "C" void kernel_launch(void* const* d_in, const int* in_sizes, int n_in,
                              void* d_out, int out_size, void* d_ws,
                              size_t ws_size, hipStream_t stream) {
  (void)in_sizes; (void)n_in; (void)out_size; (void)ws_size;
  constexpr int N = 20000, E = 320000, DC = 768, DSTY = 128, D = 512;

  const float* x_content  = (const float*)d_in[0];
  const float* x_style    = (const float*)d_in[1];
  const int*   edge_index = (const int*)d_in[2];
  // d_in[3] = edge_type (unused by the reference)
  const float* W_post = (const float*)d_in[4];
  const float* b_post = (const float*)d_in[5];
  const float* W_styl = (const float*)d_in[6];
  const float* b_styl = (const float*)d_in[7];
  const float* ggc    = (const float*)d_in[8];
  const float* W_ih   = (const float*)d_in[9];
  const float* W_hh   = (const float*)d_in[10];
  const float* b_ih   = (const float*)d_in[11];
  const float* b_hh   = (const float*)d_in[12];
  const float* W_out  = (const float*)d_in[13];
  const float* b_out  = (const float*)d_in[14];
  float* out = (float*)d_out;

  // workspace carve-out (256B aligned), ~205 MB total
  char* ws = (char*)d_ws;
  size_t off = 0;
  auto alloc = [&](size_t bytes) -> void* {
    void* p = ws + off;
    off += (bytes + 255) & ~(size_t)255;
    return p;
  };
  __bf16* xcb  = (__bf16*)alloc((size_t)N * DC * 2);
  __bf16* xsb  = (__bf16*)alloc((size_t)N * DSTY * 2);
  __bf16* Wpb  = (__bf16*)alloc((size_t)256 * DC * 2);
  __bf16* Wsb  = (__bf16*)alloc((size_t)256 * DSTY * 2);
  __bf16* ggcT = (__bf16*)alloc((size_t)3 * D * D * 2);
  __bf16* Wihb = (__bf16*)alloc((size_t)3 * D * D * 2);
  __bf16* Whhb = (__bf16*)alloc((size_t)3 * D * D * 2);
  float*  h    = (float*) alloc((size_t)N * D * 4);
  __bf16* hb   = (__bf16*)alloc((size_t)N * D * 2);
  float*  m    = (float*) alloc((size_t)N * D * 4);
  float*  agg  = (float*) alloc((size_t)N * D * 4);
  __bf16* aggb = (__bf16*)alloc((size_t)N * D * 2);

  auto cvt = [&](const float* s, __bf16* dvec, size_t n) {
    int n4 = (int)(n >> 2);
    cvt_f32_bf16<<<(n4 + 255) / 256, 256, 0, stream>>>(s, dvec, n4);
  };

  // bf16 copies of activations / weights
  cvt(x_content, xcb, (size_t)N * DC);
  cvt(x_style,   xsb, (size_t)N * DSTY);
  cvt(W_post,    Wpb, (size_t)256 * DC);
  cvt(W_styl,    Wsb, (size_t)256 * DSTY);
  cvt(W_ih,      Wihb, (size_t)3 * D * D);
  cvt(W_hh,      Whhb, (size_t)3 * D * D);
  transpose_cvt_ggc<<<(3 * D * D + 255) / 256, 256, 0, stream>>>(ggc, ggcT);

  // encoders -> h = concat(xc, xs)  [N, 512]
  dim3 blk(256);
  dim3 gridEnc((N + 127) / 128, 256 / 128);
  gemm_bf16_wT<<<gridEnc, blk, 0, stream>>>(xcb, Wpb, b_post, h, N, DC, D, 0);
  gemm_bf16_wT<<<gridEnc, blk, 0, stream>>>(xsb, Wsb, b_styl, h, N, DSTY, D, 256);

  // 3 GatedGraphConv layers
  for (int l = 0; l < 3; ++l) {
    cvt(h, hb, (size_t)N * D);
    dim3 gridM((N + 127) / 128, D / 128);
    gemm_bf16_wT<<<gridM, blk, 0, stream>>>(hb, ggcT + (size_t)l * D * D,
                                            nullptr, m, N, D, D, 0);
    int n4 = (N * D) >> 2;
    zero_f32_4<<<(n4 + 255) / 256, 256, 0, stream>>>((float4*)agg, n4);
    long tot = (long)E * 128;
    scatter_add_edges<<<(unsigned)((tot + 255) / 256), 256, 0, stream>>>(
        m, edge_index, agg, E);
    cvt(agg, aggb, (size_t)N * D);
    dim3 gridG(N / 16, D / 16);
    gru_fused<<<gridG, dim3(32), 0, stream>>>(aggb, hb, h, Wihb, Whhb,
                                              b_ih, b_hh);
  }

  // relu + linear head -> [N, 2]
  out_head<<<(N + 255) / 256, 256, 0, stream>>>(h, W_out, b_out, out, N);
}
